// ContextAwareModel_23502061043785
// MI455X (gfx1250) — matrix operand order repaired and checked by reading, not compiled, not run on hardware
//
#include <hip/hip_runtime.h>
#include <hip/hip_bf16.h>
#include <stddef.h>
#include <stdint.h>

typedef __attribute__((ext_vector_type(16))) __bf16 v16bf;
typedef __attribute__((ext_vector_type(8)))  float  v8f;
typedef __attribute__((ext_vector_type(4)))  unsigned int uint4v;
typedef __attribute__((ext_vector_type(4)))  int i4;

// address-space-qualified pointer types for the async-to-LDS builtin
typedef __attribute__((address_space(1))) i4* as1_i4p;
typedef __attribute__((address_space(3))) i4* as3_i4p;

#define LDOC 200
#define BATCH 64
#define EDIM 768
#define HDIM 384
#define GDIM 1536   // 4*H
#define CELL_ELEMS (BATCH*HDIM)   // 24576

#define CHUNK_BYTES 5120          // A 16x32 bf16 (1KB) + B 4x16x32 bf16 (4KB)
#define NBUF 3                    // triple buffer, prefetch distance 2
#define WAVES_PER_BLOCK 4

// ---------------- workspace layout (bytes) ----------------
#define OFF_WIH   ((size_t)0)                           // 4*1536*768 bf16
#define OFF_WHH   (OFF_WIH + (size_t)4*GDIM*EDIM*2)     // 4*1536*384 bf16
#define OFF_X     (OFF_WHH + (size_t)4*GDIM*HDIM*2)     // 200*64*768 bf16
#define OFF_H     (OFF_X   + (size_t)LDOC*BATCH*EDIM*2) // 2 bufs * 4 cells * 24576 bf16
#define OFF_C     (OFF_H   + (size_t)2*4*CELL_ELEMS*2)  // 4 cells * 24576 f32
#define OFF_REPS  (OFF_C   + (size_t)4*CELL_ELEMS*4)    // 200*64*768 f32

__device__ __forceinline__ float sigf(float x) {
    return 1.0f / (1.0f + __expf(-x));
}

// ---------------- async global->LDS copy (CDNA5 async path, ASYNCcnt) -------
__device__ __forceinline__ void async_b128(const void* g, void* l) {
#if __has_builtin(__builtin_amdgcn_global_load_async_to_lds_b128)
    // signature (from clang diagnostic): (v4i as1*, v4i as3*, imm offset, imm cpol)
    as1_i4p gp = (as1_i4p)(uintptr_t)g;              // AS0 and AS1 share representation
    as3_i4p lp = (as3_i4p)(unsigned)(uintptr_t)l;    // LDS offset = low 32 bits of generic
    __builtin_amdgcn_global_load_async_to_lds_b128(gp, lp, 0, 0);
#else
    unsigned loff = (unsigned)(uintptr_t)l;
    asm volatile("global_load_async_to_lds_b128 %0, %1, off"
                 :: "v"(loff), "v"(g) : "memory");
#endif
}

#if __has_builtin(__builtin_amdgcn_s_wait_asynccnt)
#define WAIT_ASYNC(n) __builtin_amdgcn_s_wait_asynccnt(n)
#else
#define WAIT_ASYNC(n) asm volatile("s_wait_asynccnt %0" :: "i"(n) : "memory")
#endif

// ---------------- LDS fragment loads (ds_load_b128 pairs) ----------------
// A-matrix 16x32 bf16 (ISA 7.12.2): lane m holds row lane&15;
// half 0 -> local K {0..7,16..23}; half 1 -> {8..15,24..31}. Chunk ld = 32.
__device__ __forceinline__ v16bf lds_a_frag(const char* buf, int lane) {
    int half = lane >> 4, row = lane & 15;
    const __bf16* p = (const __bf16*)buf + row * 32 + 8 * half;
    union { uint4v u[2]; v16bf v; } u;
    u.u[0] = *(const uint4v*)(p);
    u.u[1] = *(const uint4v*)(p + 16);
    return u.v;
}
// B-matrix 32x16: lane holds W row (= B column) lane&15; half 0 -> K {0..7,16..23},
// half 1 -> K {8..15,24..31}.
__device__ __forceinline__ v16bf lds_b_frag(const char* buf, int gate, int lane) {
    int half = lane >> 4, row = lane & 15;
    const __bf16* p = (const __bf16*)(buf + 1024 + gate * 1024) + row * 32 + 8 * half;
    union { uint4v u[2]; v16bf v; } u;
    u.u[0] = *(const uint4v*)(p);
    u.u[1] = *(const uint4v*)(p + 16);
    return u.v;
}

struct Frags { v16bf a, b0, b1, b2, b3; };

__device__ __forceinline__ void mma4(const Frags& f, v8f acc[4]) {
    acc[0] = __builtin_amdgcn_wmma_f32_16x16x32_bf16(false, f.a, false, f.b0, (short)0, acc[0], false, false);
    acc[1] = __builtin_amdgcn_wmma_f32_16x16x32_bf16(false, f.a, false, f.b1, (short)0, acc[1], false, false);
    acc[2] = __builtin_amdgcn_wmma_f32_16x16x32_bf16(false, f.a, false, f.b2, (short)0, acc[2], false, false);
    acc[3] = __builtin_amdgcn_wmma_f32_16x16x32_bf16(false, f.a, false, f.b3, (short)0, acc[3], false, false);
}

// One K-chunk's global->LDS prefetch: 2 async b128 for A, 8 for B (512B/instr).
__device__ __forceinline__ void prefetch_chunk(const __bf16* __restrict__ A, int m0,
                                               int a_ld, int a_kk,
                                               const __bf16* __restrict__ W, int n0,
                                               int w_ld, int w_kk,
                                               char* buf, int lane) {
#pragma unroll
    for (int i = 0; i < 2; ++i) {              // A tile: 16 rows x 64B
        int s = lane + 32 * i;                 // 0..63 16B-segments
        int row = s >> 2, qw = s & 3;
        async_b128(A + (size_t)(m0 + row) * a_ld + a_kk + qw * 8,
                   buf + row * 64 + qw * 16);
    }
#pragma unroll
    for (int i = 0; i < 8; ++i) {              // B tiles: 4 gates x 16 rows x 64B
        int s = lane + 32 * i;                 // 0..255
        int gate = s >> 6, row = (s >> 2) & 15, qw = s & 3;
        async_b128(W + (size_t)(gate * HDIM + n0 + row) * w_ld + w_kk + qw * 8,
                   buf + 1024 + gate * 1024 + row * 64 + qw * 16);
    }
}

struct Src { const __bf16* A; int a_ld, a_kk; const __bf16* W; int w_ld, w_kk; };

// Fully unrolled, triple-buffered, async-pipelined fused 4-gate GEMM.
template <int NCH, typename F>
__device__ __forceinline__ void gate_gemm_lds(const F& srcf, int m0, int n0,
                                              char* wavebuf, int lane, v8f acc[4]) {
    {
        Src s0 = srcf(0);
        prefetch_chunk(s0.A, m0, s0.a_ld, s0.a_kk, s0.W, n0, s0.w_ld, s0.w_kk,
                       wavebuf, lane);
        Src s1 = srcf(1);
        prefetch_chunk(s1.A, m0, s1.a_ld, s1.a_kk, s1.W, n0, s1.w_ld, s1.w_kk,
                       wavebuf + CHUNK_BYTES, lane);
    }
#pragma unroll
    for (int j = 0; j < NCH; ++j) {
        if (j + 2 < NCH) {
            Src s = srcf(j + 2);
            prefetch_chunk(s.A, m0, s.a_ld, s.a_kk, s.W, n0, s.w_ld, s.w_kk,
                           wavebuf + ((j + 2) % NBUF) * CHUNK_BYTES, lane);
            WAIT_ASYNC(20);            // chunks j+1, j+2 still in flight; j landed
        } else if (j + 1 < NCH) {
            WAIT_ASYNC(10);
        } else {
            WAIT_ASYNC(0);
        }
        const char* cbuf = wavebuf + (j % NBUF) * CHUNK_BYTES;
        Frags f;
        f.a  = lds_a_frag(cbuf, lane);
        f.b0 = lds_b_frag(cbuf, 0, lane);
        f.b1 = lds_b_frag(cbuf, 1, lane);
        f.b2 = lds_b_frag(cbuf, 2, lane);
        f.b3 = lds_b_frag(cbuf, 3, lane);
        mma4(f, acc);
    }
}

// ---------------- helper kernels ----------------
__global__ void f2bf_kernel(const float* __restrict__ src, __bf16* __restrict__ dst, int n) {
    int i = blockIdx.x * blockDim.x + threadIdx.x;
    if (i < n) dst[i] = (__bf16)src[i];
}

__global__ void embed_kernel(const int* __restrict__ ctx, const float* __restrict__ Wemb,
                             __bf16* __restrict__ X) {
    int i = blockIdx.x * blockDim.x + threadIdx.x;   // over LDOC*BATCH*EDIM
    if (i >= LDOC * BATCH * EDIM) return;
    int e  = i % EDIM;
    int tb = i / EDIM;
    int b  = tb % BATCH;
    int t  = tb / BATCH;
    int tok = ctx[b * LDOC + t];
    X[i] = (__bf16)Wemb[(size_t)tok * EDIM + e];
}

__global__ void init_state_kernel(__bf16* __restrict__ h0, float* __restrict__ c) {
    int i = blockIdx.x * blockDim.x + threadIdx.x;
    if (i < 4 * CELL_ELEMS) { h0[i] = (__bf16)0.0f; c[i] = 0.0f; }
}

// ---------------- LSTM step: layer 0 (cells 0,1) ----------------
// 48 blocks x 128 threads = 192 waves; wave = (cell d, row-tile mi, col-tile nj).
__global__ __launch_bounds__(128) void lstm_step_l0(
    const __bf16* __restrict__ Xbf, const __bf16* __restrict__ Wih,
    const __bf16* __restrict__ Whh, const float* __restrict__ bl,
    const __bf16* __restrict__ hprev, __bf16* __restrict__ hcur,
    float* __restrict__ cst, int t)
{
    __shared__ __align__(16) char smem[WAVES_PER_BLOCK * NBUF * CHUNK_BYTES];
    int lane = threadIdx.x & 31;
    char* mybuf = smem + (threadIdx.x >> 5) * (NBUF * CHUNK_BYTES);
    int wid  = blockIdx.x * WAVES_PER_BLOCK + (threadIdx.x >> 5);  // 0..191
    int d    = wid / 96;                              // cell 0/1
    int rem  = wid % 96;
    int m0   = (rem / 24) * 16;                       // batch-row tile
    int n0   = (rem % 24) * 16;                       // hidden-col tile
    int k    = d;

    const __bf16* Ax  = Xbf + (size_t)t * BATCH * EDIM;
    const __bf16* Wik = Wih + (size_t)k * GDIM * EDIM;
    const __bf16* Whk = Whh + (size_t)k * GDIM * HDIM;
    const __bf16* hp  = hprev + (size_t)k * CELL_ELEMS;

    v8f acc[4] = {};   // i, f, g, o gate tiles
    auto srcf = [&](int j) -> Src {
        if (j < 24) return Src{Ax, EDIM, 32 * j,        Wik, EDIM, 32 * j};
        return        Src{hp, HDIM, 32 * (j - 24), Whk, HDIM, 32 * (j - 24)};
    };
    gate_gemm_lds<36>(srcf, m0, n0, mybuf, lane, acc);   // K = 768 + 384

    // elementwise cell update (C/D layout: lane&15 -> N, VGPR r -> M(+8 for half 1))
    int n = n0 + (lane & 15);
    int mbase = m0 + ((lane >> 4) << 3);
    float bi = bl[k * GDIM + 0 * HDIM + n];
    float bf = bl[k * GDIM + 1 * HDIM + n];
    float bg = bl[k * GDIM + 2 * HDIM + n];
    float bo = bl[k * GDIM + 3 * HDIM + n];
    float*  cp = cst  + (size_t)k * CELL_ELEMS;
    __bf16* hc = hcur + (size_t)k * CELL_ELEMS;
#pragma unroll
    for (int r = 0; r < 8; ++r) {
        size_t idx = (size_t)(mbase + r) * HDIM + n;
        float iv = sigf(acc[0][r] + bi);
        float fv = sigf(acc[1][r] + bf);
        float gv = tanhf(acc[2][r] + bg);
        float ov = sigf(acc[3][r] + bo);
        float cn = fv * cp[idx] + iv * gv;
        cp[idx] = cn;
        hc[idx] = (__bf16)(ov * tanhf(cn));
    }
}

// ---------------- LSTM step: layer 1 (cells 2,3) ----------------
// input = concat(hcur[cell0], hcur[cell1]) -> K=768, plus recurrent K=384.
__global__ __launch_bounds__(128) void lstm_step_l1(
    const __bf16* __restrict__ Wih, const __bf16* __restrict__ Whh,
    const float* __restrict__ bl, const __bf16* __restrict__ hprev,
    __bf16* __restrict__ hcur, float* __restrict__ cst,
    float* __restrict__ reps, int t)
{
    __shared__ __align__(16) char smem[WAVES_PER_BLOCK * NBUF * CHUNK_BYTES];
    int lane = threadIdx.x & 31;
    char* mybuf = smem + (threadIdx.x >> 5) * (NBUF * CHUNK_BYTES);
    int wid  = blockIdx.x * WAVES_PER_BLOCK + (threadIdx.x >> 5);
    int d    = wid / 96;
    int rem  = wid % 96;
    int m0   = (rem / 24) * 16;
    int n0   = (rem % 24) * 16;
    int k    = 2 + d;

    const __bf16* Wik = Wih + (size_t)k * GDIM * EDIM;
    const __bf16* Whk = Whh + (size_t)k * GDIM * HDIM;
    const __bf16* hp  = hprev + (size_t)k * CELL_ELEMS;

    v8f acc[4] = {};
    auto srcf = [&](int j) -> Src {
        if (j < 12) return Src{hcur,              HDIM, 32 * j,        Wik, EDIM, 32 * j};
        if (j < 24) return Src{hcur + CELL_ELEMS, HDIM, 32 * (j - 12), Wik, EDIM, HDIM + 32 * (j - 12)};
        return        Src{hp,                HDIM, 32 * (j - 24), Whk, HDIM, 32 * (j - 24)};
    };
    gate_gemm_lds<36>(srcf, m0, n0, mybuf, lane, acc);   // K = 384+384 (concat) + 384

    int n = n0 + (lane & 15);
    int mbase = m0 + ((lane >> 4) << 3);
    float bi = bl[k * GDIM + 0 * HDIM + n];
    float bf = bl[k * GDIM + 1 * HDIM + n];
    float bg = bl[k * GDIM + 2 * HDIM + n];
    float bo = bl[k * GDIM + 3 * HDIM + n];
    float*  cp = cst  + (size_t)k * CELL_ELEMS;
    __bf16* hc = hcur + (size_t)k * CELL_ELEMS;
#pragma unroll
    for (int r = 0; r < 8; ++r) {
        int m = mbase + r;
        size_t idx = (size_t)m * HDIM + n;
        float iv = sigf(acc[0][r] + bi);
        float fv = sigf(acc[1][r] + bf);
        float gv = tanhf(acc[2][r] + bg);
        float ov = sigf(acc[3][r] + bo);
        float cn = fv * cp[idx] + iv * gv;
        cp[idx] = cn;
        float hn = ov * tanhf(cn);
        hc[idx] = (__bf16)hn;
        reps[((size_t)t * BATCH + m) * (2 * HDIM) + d * HDIM + n] = hn;
    }
}

// ---------------- final gather + classifier ----------------
// d_out layout (return order): logits [64*2] | probs [64*2] | target [64*768]
__global__ void finalize_kernel(const float* __restrict__ reps,
                                const int* __restrict__ positions,
                                const float* __restrict__ Wcls,
                                const float* __restrict__ bcls,
                                float* __restrict__ out)
{
    int b = blockIdx.x;
    int t = positions[b];
    const float* src = reps + ((size_t)t * BATCH + b) * (2 * HDIM);
    __shared__ float red0[128], red1[128];
    float s0 = 0.f, s1 = 0.f;
    for (int j = threadIdx.x; j < 2 * HDIM; j += 128) {
        float v = src[j];
        out[2 * BATCH * 2 + (size_t)b * (2 * HDIM) + j] = v;   // target
        s0 += v * Wcls[j];
        s1 += v * Wcls[2 * HDIM + j];
    }
    red0[threadIdx.x] = s0; red1[threadIdx.x] = s1;
    __syncthreads();
    for (int off = 64; off > 0; off >>= 1) {
        if ((int)threadIdx.x < off) {
            red0[threadIdx.x] += red0[threadIdx.x + off];
            red1[threadIdx.x] += red1[threadIdx.x + off];
        }
        __syncthreads();
    }
    if (threadIdx.x == 0) {
        float l0 = red0[0] + bcls[0];
        float l1 = red1[0] + bcls[1];
        out[b * 2 + 0] = l0;
        out[b * 2 + 1] = l1;
        out[BATCH * 2 + b * 2 + 0] = sigf(l0);
        out[BATCH * 2 + b * 2 + 1] = sigf(l1);
    }
}

extern "C" void kernel_launch(void* const* d_in, const int* in_sizes, int n_in,
                              void* d_out, int out_size, void* d_ws, size_t ws_size,
                              hipStream_t stream) {
    (void)in_sizes; (void)n_in; (void)out_size; (void)ws_size;
    const int*   ctx   = (const int*)d_in[0];
    const int*   pos   = (const int*)d_in[1];
    const float* Wemb  = (const float*)d_in[2];
    const float* Wih_f = (const float*)d_in[3];
    const float* Whh_f = (const float*)d_in[4];
    const float* bl    = (const float*)d_in[5];
    const float* Wcls  = (const float*)d_in[6];
    const float* bcls  = (const float*)d_in[7];
    float* out = (float*)d_out;

    char* ws = (char*)d_ws;
    __bf16* Wih_bf = (__bf16*)(ws + OFF_WIH);
    __bf16* Whh_bf = (__bf16*)(ws + OFF_WHH);
    __bf16* Xbf    = (__bf16*)(ws + OFF_X);
    __bf16* hbuf   = (__bf16*)(ws + OFF_H);    // [2][4][64*384]
    float*  cst    = (float*) (ws + OFF_C);    // [4][64*384]
    float*  reps   = (float*) (ws + OFF_REPS); // [200][64][768]

    {
        int n = 4 * GDIM * EDIM;
        f2bf_kernel<<<(n + 255) / 256, 256, 0, stream>>>(Wih_f, Wih_bf, n);
    }
    {
        int n = 4 * GDIM * HDIM;
        f2bf_kernel<<<(n + 255) / 256, 256, 0, stream>>>(Whh_f, Whh_bf, n);
    }
    {
        int n = LDOC * BATCH * EDIM;
        embed_kernel<<<(n + 255) / 256, 256, 0, stream>>>(ctx, Wemb, Xbf);
    }
    {
        int n = 4 * CELL_ELEMS;
        init_state_kernel<<<(n + 255) / 256, 256, 0, stream>>>(hbuf, cst);
    }

    for (int t = 0; t < LDOC; ++t) {
        __bf16* hprev = hbuf + (size_t)(t & 1)       * 4 * CELL_ELEMS;
        __bf16* hcur  = hbuf + (size_t)((t + 1) & 1) * 4 * CELL_ELEMS;
        lstm_step_l0<<<48, 128, 0, stream>>>(Xbf, Wih_bf, Whh_bf, bl, hprev, hcur, cst, t);
        lstm_step_l1<<<48, 128, 0, stream>>>(Wih_bf, Whh_bf, bl, hprev, hcur, cst, reps, t);
    }

    finalize_kernel<<<BATCH, 128, 0, stream>>>(reps, pos, Wcls, bcls, out);
}
